// SupervisedContrastiveLoss_20598663151870
// MI455X (gfx1250) — matrix operand order repaired and checked by reading, not compiled, not run on hardware
//
#include <hip/hip_runtime.h>
#include <hip/hip_bf16.h>

typedef __attribute__((ext_vector_type(16))) _Float16 v16h;
typedef __attribute__((ext_vector_type(8)))  _Float16 v8h;
typedef __attribute__((ext_vector_type(8)))  float    v8f;

#define N      8192
#define DIM    512
#define NCHUNK (N / 32)          // 256 32-wide column chunks
#define NRBLK  (N / 32)          // 256 32-row blocks
#define INV_T  (1.0f / 0.07f)    // analytic row max of logits (normalized feats)
#define EPSV   1e-6f

// ---------------------------------------------------------------------------
// Kernel 1: fp32 -> f16 copy + LDS-tiled transpose (both sides coalesced).
// Grid: 4096 blocks, each handles a 32(i) x 32(k) tile.
// ---------------------------------------------------------------------------
__global__ __launch_bounds__(256)
void scl_convert_kernel(const float* __restrict__ f,
                        _Float16* __restrict__ f16,
                        _Float16* __restrict__ ft16) {
  __shared__ _Float16 tile[32][33];          // +1 pad: no bank conflicts
  const int iTile = blockIdx.x >> 4;         // 256 i-tiles
  const int kTile = blockIdx.x & 15;         // 16  k-tiles
  const int i0 = iTile * 32, k0 = kTile * 32;
  const int c  = threadIdx.x & 31;
  const int r0 = threadIdx.x >> 5;           // 0..7

  #pragma unroll
  for (int rr = 0; rr < 32; rr += 8) {
    int r = r0 + rr;
    _Float16 h = (_Float16)f[(size_t)(i0 + r) * DIM + k0 + c];
    f16[(size_t)(i0 + r) * DIM + k0 + c] = h;          // coalesced over c
    tile[r][c] = h;
  }
  __syncthreads();
  #pragma unroll
  for (int rr = 0; rr < 32; rr += 8) {
    int r = r0 + rr;
    ft16[(size_t)(k0 + r) * N + i0 + c] = tile[c][r];  // coalesced over c
  }
}

// ---------------------------------------------------------------------------
// Kernel 2: fused Gram-matrix (WMMA f16) + row statistics.
// One block = 32 rows (2 row-tiles); 8 waves stride over 256 32-col chunks.
// Each wave keeps 4 accumulators (2 row-tiles x 2 col-tiles): every A
// fragment feeds 2 WMMAs and every B fragment feeds 2 WMMAs -> 2 b128
// loads per WMMA, and total L2 B-traffic is halved vs 16-row blocks.
// Per row i (fixed max M = 1/T, exact for L2-normalized features):
//   S = sum_{j!=i} exp(logits[i,j] - 1/T)
//   P = sum_j mask[i,j] * logits[i,j]
//   C = sum_j mask[i,j]
// sum(mask*log_prob) = P - C/T - C*log(S+eps);  term = that / (C+eps).
// ---------------------------------------------------------------------------
__global__ __launch_bounds__(256)
void scl_main_kernel(const _Float16* __restrict__ F16,
                     const _Float16* __restrict__ FT16,
                     const int* __restrict__ labels,
                     float* __restrict__ blockTerms) {
  __shared__ float part[8][32][3];   // [wave][row-in-block][{S,P,C}]
  __shared__ float red[32];

  const int tid   = threadIdx.x;
  const int wave  = tid >> 5;
  const int lane  = tid & 31;
  const int l16   = lane & 15;
  const int hhalf = lane >> 4;       // 0: lanes 0-15, 1: lanes 16-31
  const int rowBase = blockIdx.x * 32;

  // A-fragments: lane holds row (lane%16) of each row-tile; K-base by half.
  const _Float16* Arow0 = F16 + (size_t)(rowBase + l16) * DIM;
  const _Float16* Arow1 = F16 + (size_t)(rowBase + 16 + l16) * DIM;
  const int kb = hhalf ? 8 : 0;

  // D-tile rows owned by this lane: base + r + 8*hhalf  (r = vgpr index)
  int myLab0[8], myLab1[8];
  #pragma unroll
  for (int r = 0; r < 8; ++r) {
    myLab0[r] = labels[rowBase + r + 8 * hhalf];
    myLab1[r] = labels[rowBase + 16 + r + 8 * hhalf];
  }

  float S0[8], P0[8], C0[8], S1[8], P1[8], C1[8];
  #pragma unroll
  for (int r = 0; r < 8; ++r) {
    S0[r] = 0.f; P0[r] = 0.f; C0[r] = 0.f;
    S1[r] = 0.f; P1[r] = 0.f; C1[r] = 0.f;
  }

  for (int t = wave; t < NCHUNK; t += 8) {
    const int colBase = t * 32;
    const int nGlob0  = colBase + l16;        // cols for this lane's D values
    const int nGlob1  = nGlob0 + 16;
    const int colLab0 = labels[nGlob0];
    const int colLab1 = labels[nGlob1];

    v8f acc00 = {}, acc01 = {}, acc10 = {}, acc11 = {};
    #pragma unroll 2
    for (int ks = 0; ks < DIM; ks += 32) {
      // A 16x32 f16 fragments (each reused for 2 WMMAs).
      v8h a0lo = *(const v8h*)(Arow0 + ks + kb);
      v8h a0hi = *(const v8h*)(Arow0 + ks + kb + 16);
      v16h a0 = __builtin_shufflevector(a0lo, a0hi,
                0, 1, 2, 3, 4, 5, 6, 7, 8, 9, 10, 11, 12, 13, 14, 15);
      v8h a1lo = *(const v8h*)(Arow1 + ks + kb);
      v8h a1hi = *(const v8h*)(Arow1 + ks + kb + 16);
      v16h a1 = __builtin_shufflevector(a1lo, a1hi,
                0, 1, 2, 3, 4, 5, 6, 7, 8, 9, 10, 11, 12, 13, 14, 15);
      // B 32x16 fragments: lane holds column (lane%16) of each col-tile,
      // K-base by half (same K order as A; each reused for 2 WMMAs).
      const _Float16* Bc0 = F16 + (size_t)(colBase + l16) * DIM + ks + kb;
      const _Float16* Bc1 = Bc0 + (size_t)16 * DIM;
      v16h b0 = __builtin_shufflevector(*(const v8h*)(Bc0), *(const v8h*)(Bc0 + 16),
                0, 1, 2, 3, 4, 5, 6, 7, 8, 9, 10, 11, 12, 13, 14, 15);
      v16h b1 = __builtin_shufflevector(*(const v8h*)(Bc1), *(const v8h*)(Bc1 + 16),
                0, 1, 2, 3, 4, 5, 6, 7, 8, 9, 10, 11, 12, 13, 14, 15);
      acc00 = __builtin_amdgcn_wmma_f32_16x16x32_f16(false, a0, false, b0,
                                                     (short)0, acc00, false, false);
      acc01 = __builtin_amdgcn_wmma_f32_16x16x32_f16(false, a0, false, b1,
                                                     (short)0, acc01, false, false);
      acc10 = __builtin_amdgcn_wmma_f32_16x16x32_f16(false, a1, false, b0,
                                                     (short)0, acc10, false, false);
      acc11 = __builtin_amdgcn_wmma_f32_16x16x32_f16(false, a1, false, b1,
                                                     (short)0, acc11, false, false);
    }

    // Fold the four 16x16 f32 D tiles into per-row running stats.
    #pragma unroll
    for (int r = 0; r < 8; ++r) {
      const int mGlob0 = rowBase + r + 8 * hhalf;        // row-tile 0
      const int mGlob1 = mGlob0 + 16;                    // row-tile 1
      float l00 = acc00[r] * INV_T;
      float l01 = acc01[r] * INV_T;
      float l10 = acc10[r] * INV_T;
      float l11 = acc11[r] * INV_T;
      float e00 = (nGlob0 == mGlob0) ? 0.f : __expf(l00 - INV_T);
      float e01 = (nGlob1 == mGlob0) ? 0.f : __expf(l01 - INV_T);
      float e10 = (nGlob0 == mGlob1) ? 0.f : __expf(l10 - INV_T);
      float e11 = (nGlob1 == mGlob1) ? 0.f : __expf(l11 - INV_T);
      S0[r] += e00 + e01;
      S1[r] += e10 + e11;
      if ((colLab0 == myLab0[r]) && (nGlob0 != mGlob0)) { P0[r] += l00; C0[r] += 1.f; }
      if ((colLab1 == myLab0[r]) && (nGlob1 != mGlob0)) { P0[r] += l01; C0[r] += 1.f; }
      if ((colLab0 == myLab1[r]) && (nGlob0 != mGlob1)) { P1[r] += l10; C1[r] += 1.f; }
      if ((colLab1 == myLab1[r]) && (nGlob1 != mGlob1)) { P1[r] += l11; C1[r] += 1.f; }
    }
  }

  // Sum across the 16 lanes of each half-wave (N dimension of the tiles).
  #pragma unroll
  for (int r = 0; r < 8; ++r) {
    #pragma unroll
    for (int m = 1; m <= 8; m <<= 1) {
      S0[r] += __shfl_xor(S0[r], m, 32);
      P0[r] += __shfl_xor(P0[r], m, 32);
      C0[r] += __shfl_xor(C0[r], m, 32);
      S1[r] += __shfl_xor(S1[r], m, 32);
      P1[r] += __shfl_xor(P1[r], m, 32);
      C1[r] += __shfl_xor(C1[r], m, 32);
    }
  }

  if (l16 == 0) {
    #pragma unroll
    for (int r = 0; r < 8; ++r) {
      int rowLocal = r + 8 * hhalf;
      part[wave][rowLocal][0]      = S0[r];
      part[wave][rowLocal][1]      = P0[r];
      part[wave][rowLocal][2]      = C0[r];
      part[wave][rowLocal + 16][0] = S1[r];
      part[wave][rowLocal + 16][1] = P1[r];
      part[wave][rowLocal + 16][2] = C1[r];
    }
  }
  __syncthreads();

  // Merge the 8 wave-partials per row, compute per-row loss term.
  if (tid < 32) {
    float Sv = 0.f, Pv = 0.f, Cv = 0.f;
    #pragma unroll
    for (int w = 0; w < 8; ++w) {
      Sv += part[w][tid][0];
      Pv += part[w][tid][1];
      Cv += part[w][tid][2];
    }
    red[tid] = (Pv - Cv * INV_T - Cv * __logf(Sv + EPSV)) / (Cv + EPSV);
  }
  __syncthreads();

  if (tid == 0) {
    float s = 0.f;
    #pragma unroll
    for (int i = 0; i < 32; ++i) s += red[i];
    blockTerms[blockIdx.x] = s;   // deterministic: no atomics
  }
}

// ---------------------------------------------------------------------------
// Kernel 3: reduce 256 block partials, write  -mean  to d_out.
// ---------------------------------------------------------------------------
__global__ void scl_finalize_kernel(const float* __restrict__ blockTerms,
                                    float* __restrict__ out) {
  __shared__ float buf[256];
  int tid = threadIdx.x;
  buf[tid] = blockTerms[tid];
  __syncthreads();
  for (int s = 128; s > 0; s >>= 1) {
    if (tid < s) buf[tid] += buf[tid + s];
    __syncthreads();
  }
  if (tid == 0) out[0] = -buf[0] / (float)N;
}

// ---------------------------------------------------------------------------
extern "C" void kernel_launch(void* const* d_in, const int* in_sizes, int n_in,
                              void* d_out, int out_size, void* d_ws, size_t ws_size,
                              hipStream_t stream) {
  const float* feats  = (const float*)d_in[0];
  const int*   labels = (const int*)d_in[1];

  // Workspace layout:
  //   [0, 1KB)                 : 256 float block partials
  //   [4KB, 4KB+8MB)           : F16  (8192 x 512 f16, row-major)
  //   [4KB+8MB, 4KB+16MB)      : FT16 (512 x 8192 f16, transposed)
  char* ws = (char*)d_ws;
  float*    blockTerms = (float*)ws;
  _Float16* F16  = (_Float16*)(ws + 4096);
  _Float16* FT16 = (_Float16*)(ws + 4096 + (size_t)N * DIM * sizeof(_Float16));

  scl_convert_kernel<<<(N / 32) * (DIM / 32), 256, 0, stream>>>(feats, F16, FT16);
  scl_main_kernel<<<NRBLK, 256, 0, stream>>>(F16, FT16, labels, blockTerms);
  scl_finalize_kernel<<<1, 256, 0, stream>>>(blockTerms, (float*)d_out);
}
